// SPELayer_28939489641145
// MI455X (gfx1250) — matrix operand order repaired and checked by reading, not compiled, not run on hardware
//
#include <hip/hip_runtime.h>

// ---------------------------------------------------------------------------
// SPE distance kernel for MI455X (gfx1250).
// Heavy math: spe[b,i,(j,d)] = V(nxn) @ C(nx32n), C[e,(j,d)] = V[j,e]*phi[e,d]
// ~16 GFLOP total -> bf16 V_WMMA_F32_16X16X32_BF16; output 156MB -> ~6.7us HBM.
// C tiles are built on the fly: V rows and phi^T rows are both contiguous in e,
// product done with v_pk_mul_bf16, staged through double-buffered LDS.
// ---------------------------------------------------------------------------

typedef unsigned short u16;
typedef unsigned int   u32;
typedef __attribute__((ext_vector_type(16))) __bf16 bf16x16;
typedef __attribute__((ext_vector_type(8)))  __bf16 bf16x8;
typedef __attribute__((ext_vector_type(8)))  float  f32x8;

union BFrag { bf16x16 v; uint4 q[2]; };
union HQ    { bf16x8  h; uint4 q; };

static __device__ __forceinline__ u16 f2bf(float f) {
  union { float f; u32 u; } a; a.f = f;
  u32 r = a.u + 0x7FFFu + ((a.u >> 16) & 1u);   // round-to-nearest-even
  return (u16)(r >> 16);
}
static __device__ __forceinline__ float bf2f(u16 h) {
  union { u32 u; float f; } a; a.u = ((u32)h) << 16;
  return a.f;
}

#define NB 32       // batch
#define NMAX 256
#define NPAD 40     // u16 per column per 32-K chunk (80B: conflict-free b128 reads)

// ---------------- meta: [0..31]=counts  [32..64]=node_off  [65..97]=pair_off
__global__ void k_offsets(const int* __restrict__ batch, int n_tot, int* __restrict__ meta) {
  __shared__ int sc[NB];
  int t = threadIdx.x;
  if (t < NB) sc[t] = 0;
  __syncthreads();
  for (int i = t; i < n_tot; i += blockDim.x) atomicAdd(&sc[batch[i]], 1);
  __syncthreads();
  if (t == 0) {
    int no = 0, po = 0;
    for (int b = 0; b < NB; b++) {
      meta[b] = sc[b]; meta[32 + b] = no; meta[65 + b] = po;
      no += sc[b]; po += sc[b] * sc[b];
    }
    meta[64] = no; meta[97] = po;
  }
}

// ---------------- dense bf16 V (row-major) + dense eigvals -------------------
__global__ void k_unpack(const float* __restrict__ eigvals, const float* __restrict__ evec,
                         const int* __restrict__ meta,
                         u16* __restrict__ Vbf, float* __restrict__ evd) {
  int b = blockIdx.x, i = blockIdx.y, j = threadIdx.x;
  int n = meta[b];
  float v = 0.0f;
  if (i < n && j < n) v = evec[meta[65 + b] + i * n + j];
  Vbf[((size_t)b << 16) + (size_t)i * NMAX + j] = f2bf(v);
  if (i == 0) evd[b * NMAX + j] = (j < n) ? eigvals[meta[32 + b] + j] : 0.0f;
}

// ---------------- MLP stage 0: g0[c], pooled1[c], h1[c] per graph ------------
__global__ void k_mlp0(const float* __restrict__ evd,
                       const float* __restrict__ W0, const float* __restrict__ U0, const float* __restrict__ b0,
                       const float* __restrict__ U1, const float* __restrict__ b1,
                       float* __restrict__ G) {          // G[b*192]: g0|pooled1|h1
  int b = blockIdx.x, c = threadIdx.x;
  __shared__ float p1s[64];
  const float* ev = &evd[b * NMAX];
  float pooled0 = 0.0f;
  for (int nn = 0; nn < NMAX; nn++) pooled0 += ev[nn];
  float g0 = pooled0 * U0[c] + b0[c];
  float w0 = W0[c];
  float p1 = 0.0f;
  for (int nn = 0; nn < NMAX; nn++) {
    float x = ev[nn] * w0 + g0;
    p1 += (x > 0.0f) ? x : 0.0f;               // relu(layer0), pooled over ALL 256 rows
  }
  p1s[c] = p1;
  float* o = &G[b * 192];
  o[c] = g0; o[64 + c] = p1;
  __syncthreads();
  float h1 = b1[c];
  for (int ci = 0; ci < 64; ci++) h1 += p1s[ci] * U1[ci * 64 + c];
  o[128 + c] = h1;
}

// ---------------- MLP stage 1: x2r[b][n][c] = relu(layer1) -------------------
__global__ void k_mlp1(const float* __restrict__ evd, const float* __restrict__ W0,
                       const float* __restrict__ W1, const float* __restrict__ G,
                       float* __restrict__ x2r) {
  __shared__ float W1s[64 * 64];
  int b = blockIdx.x;
  int n = blockIdx.y * 64 + threadIdx.x;
  for (int k = threadIdx.x; k < 64 * 64; k += 64) W1s[k] = W1[k];
  __syncthreads();
  const float* g = &G[b * 192];
  float ev = evd[b * NMAX + n];
  float r1[64];
  for (int c = 0; c < 64; c++) {
    float x = ev * W0[c] + g[c];
    r1[c] = (x > 0.0f) ? x : 0.0f;
  }
  float* o = &x2r[((size_t)b * NMAX + n) * 64];
  for (int co = 0; co < 64; co++) {
    float acc = g[128 + co];
    for (int ci = 0; ci < 64; ci++) acc += r1[ci] * W1s[ci * 64 + co];
    o[co] = (acc > 0.0f) ? acc : 0.0f;
  }
}

// ---------------- MLP stage 2: h2[b][d] = b2 + pooled2 @ U2 ------------------
__global__ void k_mlp2(const float* __restrict__ x2r, const float* __restrict__ U2,
                       const float* __restrict__ b2, float* __restrict__ h2) {
  int b = blockIdx.x, c = threadIdx.x;
  __shared__ float p2s[64];
  const float* X = &x2r[(size_t)b * NMAX * 64];
  float p2 = 0.0f;
  for (int nn = 0; nn < NMAX; nn++) p2 += X[nn * 64 + c];
  p2s[c] = p2;
  __syncthreads();
  if (c < 32) {
    float h = b2[c];
    for (int ci = 0; ci < 64; ci++) h += p2s[ci] * U2[ci * 32 + c];
    h2[b * 32 + c] = h;
  }
}

// ---------------- phiT[b][d][e] (bf16, d-major: contiguous in e) -------------
__global__ void k_phi(const float* __restrict__ x2r, const float* __restrict__ W2,
                      const float* __restrict__ h2, u16* __restrict__ phiT) {
  int b = blockIdx.x;
  int n = blockIdx.y * 8 + (threadIdx.x >> 5);
  int d = threadIdx.x & 31;
  const float* X = &x2r[((size_t)b * NMAX + n) * 64];
  float acc = h2[b * 32 + d];
  for (int c = 0; c < 64; c++) acc += X[c] * W2[c * 32 + d];
  phiT[((size_t)b * 32 + d) * NMAX + n] = f2bf(acc);
}

// ---------------- gd[b][i][d] = sum_e V[i,e]^2 * phi[e,d] (bf16 inputs) ------
__global__ void k_gd(const u16* __restrict__ Vbf, const u16* __restrict__ phiT,
                     float* __restrict__ gd) {
  int b = blockIdx.x;
  int i = blockIdx.y * 8 + (threadIdx.x >> 5);
  int d = threadIdx.x & 31;
  const uint4* vr = (const uint4*)&Vbf[((size_t)b << 16) + (size_t)i * NMAX];
  const uint4* pr = (const uint4*)&phiT[((size_t)b * 32 + d) * NMAX];
  float acc = 0.0f;
  for (int e8 = 0; e8 < NMAX / 8; e8++) {
    HQ v, p; v.q = vr[e8]; p.q = pr[e8];
    const u16* vu = (const u16*)&v;
    const u16* pu = (const u16*)&p;
    #pragma unroll
    for (int k = 0; k < 8; k++) {
      float x = bf2f(vu[k]);
      acc += x * x * bf2f(pu[k]);
    }
  }
  gd[((size_t)b * NMAX + i) * 32 + d] = acc;
}

// ---------------- main WMMA kernel -------------------------------------------
// grid(b, jb): 8 waves; wave w owns column j=jb*8+w (32 d's = 2 N-tiles),
// M strip of 128 i's per pass, K streamed 32-wide through double-buffered LDS.
__global__ __launch_bounds__(256) void k_spe(
    const u16* __restrict__ Vbf, const u16* __restrict__ phiT,
    const float* __restrict__ gd, const int* __restrict__ meta,
    float* __restrict__ out) {
  __shared__ u16 Clds[2][256 * NPAD];
  const int b = blockIdx.x, jb = blockIdx.y;
  const int n = meta[b];
  const int j0 = jb * 8;
  if (j0 >= n) return;                         // uniform: EXEC stays all-ones
  const int t = threadIdx.x;
  const int w = __builtin_amdgcn_readfirstlane(t >> 5);  // wave id (uniform)
  const int lane = t & 31;
  const bool hi = lane >= 16;
  const int npad = (n + 31) & ~31;
  const u16* V_b = Vbf + ((size_t)b << 16);
  const int myj = j0 + w;
  const long pairoff = meta[65 + b];

  // C-tile build sources: thread t -> column t (j = j0 + w, d = lane)
  // V[j, e]     = Vbf row j      (contiguous in e, wave-uniform row)
  // phi[e, d]   = phiT row d     (contiguous in e)
  const u16* vrow = V_b + (size_t)myj * NMAX;
  const u16* prow = phiT + ((size_t)b * 32 + lane) * NMAX;
  u16* mycol = &Clds[0][0] + t * NPAD;

  const int colb = (w << 5) + (hi ? lane - 16 : lane);  // B-frag col, d-half 0
  const int row  = lane & 15;                           // A-frag row within tile

  auto build = [&](int e0, int pb) {
    const uint4* pv = (const uint4*)(vrow + e0);
    const uint4* pp = (const uint4*)(prow + e0);
    uint4* dst = (uint4*)(mycol + pb * (256 * NPAD));
    #pragma unroll
    for (int k = 0; k < 4; k++) {
      HQ v, p, c;
      v.q = pv[k]; p.q = pp[k];
      c.h = v.h * p.h;                         // v_pk_mul_bf16 x4
      dst[k] = c.q;
    }
  };

  for (int i0 = 0; i0 < n; i0 += 128) {
    f32x8 acc[8][2];
    #pragma unroll
    for (int a = 0; a < 8; a++)
      #pragma unroll
      for (int q = 0; q < 2; q++)
        acc[a][q] = (f32x8){0.f, 0.f, 0.f, 0.f, 0.f, 0.f, 0.f, 0.f};

    __syncthreads();                           // Clds reuse across i0 passes
    build(0, 0);                               // prologue: chunk 0 -> buffer 0

    int p = 0;
    for (int e0 = 0; e0 < npad; e0 += 32) {
      __syncthreads();                         // buffer p ready for everyone
      if (e0 + 32 < npad) build(e0 + 32, p ^ 1);  // overlaps with WMMAs below

      // B fragments (2 N-tiles of this wave): lane<16 -> K 0..15, hi -> K 16..31
      BFrag bf0, bf1;
      {
        const uint4* pc0 = (const uint4*)&Clds[p][(colb)      * NPAD + (hi ? 16 : 0)];
        const uint4* pc1 = (const uint4*)&Clds[p][(colb + 16) * NPAD + (hi ? 16 : 0)];
        bf0.q[0] = pc0[0]; bf0.q[1] = pc0[1];
        bf1.q[0] = pc1[0]; bf1.q[1] = pc1[1];
      }
      const int ecol = e0 + (hi ? 16 : 0);     // A layout: lo lanes K0-15, hi K16-31
      #pragma unroll
      for (int it = 0; it < 8; it++) {
        const uint4* pa = (const uint4*)&V_b[(size_t)(i0 + it * 16 + row) * NMAX + ecol];
        if (e0 + 32 < npad)
          __builtin_prefetch((const void*)(pa + 4), 0, 3);   // next K strip
        BFrag af; af.q[0] = pa[0]; af.q[1] = pa[1];
        acc[it][0] = __builtin_amdgcn_wmma_f32_16x16x32_bf16(
            false, af.v, false, bf0.v, (short)0, acc[it][0], false, false);
        acc[it][1] = __builtin_amdgcn_wmma_f32_16x16x32_bf16(
            false, af.v, false, bf1.v, (short)0, acc[it][1], false, false);
      }
      p ^= 1;
    }

    // epilogue: dist = gd_i + gd_j - 2*spe, scattered to ragged output
    if (myj < n) {
      const float* gdj = gd + ((size_t)b * NMAX + myj) * 32;
      #pragma unroll
      for (int q = 0; q < 2; q++) {
        int d = q * 16 + (lane & 15);          // D layout: lane = column
        float gj = gdj[d];
        #pragma unroll
        for (int it = 0; it < 8; it++) {
          int ib = i0 + it * 16 + (hi ? 8 : 0);
          #pragma unroll
          for (int r = 0; r < 8; r++) {        // D layout: VGPR r = row (+8 for hi lanes)
            int i = ib + r;
            if (i < n) {
              float gi = gd[((size_t)b * NMAX + i) * 32 + d];
              out[(pairoff + (long)i * n + myj) * 32 + d] = gi + gj - 2.0f * acc[it][q][r];
            }
          }
        }
      }
    }
  }
}

// ---------------------------------------------------------------------------
extern "C" void kernel_launch(void* const* d_in, const int* in_sizes, int n_in,
                              void* d_out, int out_size, void* d_ws, size_t ws_size,
                              hipStream_t stream) {
  const float* eigvals = (const float*)d_in[0];
  const float* evec    = (const float*)d_in[1];
  const int*   batch   = (const int*)d_in[2];
  const float* W0 = (const float*)d_in[3];
  const float* U0 = (const float*)d_in[4];
  const float* b0 = (const float*)d_in[5];
  const float* W1 = (const float*)d_in[6];
  const float* U1 = (const float*)d_in[7];
  const float* b1 = (const float*)d_in[8];
  const float* W2 = (const float*)d_in[9];
  const float* U2 = (const float*)d_in[10];
  const float* b2 = (const float*)d_in[11];
  const int n_tot = in_sizes[0];
  float* out = (float*)d_out;

  // workspace carve-out (~7.6 MB), 256B aligned slices
  char* w = (char*)d_ws;
  auto alloc = [&](size_t bytes) -> char* {
    char* p = w; w += (bytes + 255) & ~(size_t)255; return p;
  };
  int*   meta  = (int*)alloc(98 * sizeof(int));
  float* evd   = (float*)alloc((size_t)NB * NMAX * 4);
  u16*   Vbf   = (u16*)alloc((size_t)NB * NMAX * NMAX * 2);
  u16*   phiT  = (u16*)alloc((size_t)NB * NMAX * 32 * 2);
  float* gd    = (float*)alloc((size_t)NB * NMAX * 32 * 4);
  float* G     = (float*)alloc((size_t)NB * 192 * 4);
  float* x2r   = (float*)alloc((size_t)NB * NMAX * 64 * 4);
  float* h2    = (float*)alloc((size_t)NB * 32 * 4);
  (void)ws_size; (void)n_in; (void)out_size;

  k_offsets<<<1, 256, 0, stream>>>(batch, n_tot, meta);
  k_unpack <<<dim3(NB, NMAX), NMAX, 0, stream>>>(eigvals, evec, meta, Vbf, evd);
  k_mlp0   <<<NB, 64, 0, stream>>>(evd, W0, U0, b0, U1, b1, G);
  k_mlp1   <<<dim3(NB, 4), 64, 0, stream>>>(evd, W0, W1, G, x2r);
  k_mlp2   <<<NB, 64, 0, stream>>>(x2r, U2, b2, h2);
  k_phi    <<<dim3(NB, 32), 256, 0, stream>>>(x2r, W2, h2, phiT);
  k_gd     <<<dim3(NB, 32), 256, 0, stream>>>(Vbf, phiT, gd);
  k_spe    <<<dim3(NB, 32), 256, 0, stream>>>(Vbf, phiT, gd, meta, out);
}
